// DiffusionRetrievalGNN_87084756894022
// MI455X (gfx1250) — matrix operand order repaired and checked by reading, not compiled, not run on hardware
//
#include <hip/hip_runtime.h>
#include <hip/hip_bf16.h>

#define H 128
#define NT 20000
#define NC 100000
#define E_TC 100000
#define E_SIM 800000
#define E_TS 640000
#define K_APPNP 10
#define ALPHA 0.2f

typedef __attribute__((ext_vector_type(16))) __bf16 v16bf;
typedef __attribute__((ext_vector_type(8)))  __bf16 v8bf;
typedef __attribute__((ext_vector_type(4)))  __bf16 v4bf;
typedef __attribute__((ext_vector_type(8)))  float  v8f;

#define LDA 40  // padded bf16 LDS stride for the 32x32 A tile

// ---------------------------------------------------------------------------
// Weight prep: out[n*K + k] = bf16( sum_i Wi[k*128 + n] )   (transpose + sum)
// ---------------------------------------------------------------------------
__global__ __launch_bounds__(256) void wtprep_kernel(
    const float* __restrict__ W0, const float* __restrict__ W1,
    const float* __restrict__ W2, const float* __restrict__ W3,
    int K, __bf16* __restrict__ out) {
  int i = blockIdx.x * blockDim.x + threadIdx.x;
  if (i >= K * H) return;
  int n = i / K;
  int k = i - n * K;
  float v = W0[(size_t)k * H + n];
  if (W1) v += W1[(size_t)k * H + n];
  if (W2) v += W2[(size_t)k * H + n];
  if (W3) v += W3[(size_t)k * H + n];
  out[i] = (__bf16)v;
}

__global__ void bias4_kernel(const float* __restrict__ b0, const float* __restrict__ b1,
                             const float* __restrict__ b2, const float* __restrict__ b3,
                             float* __restrict__ out) {
  int h = threadIdx.x;
  float v = b0[h];
  if (b1) v += b1[h];
  if (b2) v += b2[h];
  if (b3) v += b3[h];
  out[h] = v;
}

// ---------------------------------------------------------------------------
// C[M,128] = A[M,K] (f32) @ W (Wt = bf16, transposed [128][K]) + bias
// Block: 256 thr (8 waves), tile 32(M) x 128(N), K-step 32.
// wave w: mt = w>>2 (M sub-tile), np = w&3 -> N tiles {2np, 2np+1}
// ---------------------------------------------------------------------------
__global__ __launch_bounds__(256) void gemm_wmma_kernel(
    const float* __restrict__ A, const __bf16* __restrict__ Wt,
    float* __restrict__ C, int M, int K, const float* __restrict__ bias) {
  __shared__ __attribute__((aligned(16))) __bf16 As[32 * LDA];
  const int tid   = threadIdx.x;
  const int wave  = tid >> 5;
  const int lane  = tid & 31;
  const int mt    = wave >> 2;         // 0..1
  const int np    = wave & 3;          // 0..3
  const int mBlk  = blockIdx.x * 32;
  const int lrow  = lane & 15;
  const int kbA   = (lane < 16) ? 0 : 16;   // A-matrix K base per half-wave
  const int kbB   = (lane < 16) ? 0 : 16;   // B-matrix K base per half-wave
  const int cn0   = np * 32;
  const int cn1   = np * 32 + 16;

  v8f c0 = {};
  v8f c1 = {};

  // staging indices: 8 threads per row, 4 consecutive floats each
  const int sR = tid >> 3;
  const int sK = (tid & 7) * 4;

  for (int kk = 0; kk < K; kk += 32) {
    __syncthreads();
    {
      const float4 v = *(const float4*)(A + (size_t)(mBlk + sR) * K + kk + sK);
      v4bf t;
      t[0] = (__bf16)v.x; t[1] = (__bf16)v.y; t[2] = (__bf16)v.z; t[3] = (__bf16)v.w;
      *(v4bf*)(&As[sR * LDA + sK]) = t;
    }
    __syncthreads();

    union { v16bf v; v8bf h[2]; } af, b0, b1;
    const __bf16* ap = &As[(mt * 16 + lrow) * LDA + kbA];
    af.h[0] = *(const v8bf*)ap;         // K = kbA .. kbA+7
    af.h[1] = *(const v8bf*)(ap + 8);   // K = kbA+8 .. kbA+15
    const __bf16* bp0 = Wt + (size_t)(cn0 + lrow) * K + kk + kbB;
    b0.h[0] = *(const v8bf*)bp0;
    b0.h[1] = *(const v8bf*)(bp0 + 8);
    const __bf16* bp1 = Wt + (size_t)(cn1 + lrow) * K + kk + kbB;
    b1.h[0] = *(const v8bf*)bp1;
    b1.h[1] = *(const v8bf*)(bp1 + 8);

    c0 = __builtin_amdgcn_wmma_f32_16x16x32_bf16(false, af.v, false, b0.v,
                                                 (short)0, c0, false, false);
    c1 = __builtin_amdgcn_wmma_f32_16x16x32_bf16(false, af.v, false, b1.v,
                                                 (short)0, c1, false, false);
  }

  const int rbase = mBlk + mt * 16 + ((lane >> 4) & 1) * 8;
  {
    const int ccol = cn0 + lrow;
    const float bv = bias ? bias[ccol] : 0.f;
    #pragma unroll
    for (int r = 0; r < 8; ++r)
      C[(size_t)(rbase + r) * H + ccol] = c0[r] + bv;
  }
  {
    const int ccol = cn1 + lrow;
    const float bv = bias ? bias[ccol] : 0.f;
    #pragma unroll
    for (int r = 0; r < 8; ++r)
      C[(size_t)(rbase + r) * H + ccol] = c1[r] + bv;
  }
}

// ---------------------------------------------------------------------------
// Degree / edge kernels
// ---------------------------------------------------------------------------
__global__ __launch_bounds__(256) void count_deg_kernel(const int* __restrict__ dst,
                                                        int E, float* __restrict__ deg) {
  int e = blockIdx.x * blockDim.x + threadIdx.x;
  if (e < E) atomicAdd(&deg[dst[e]], 1.0f);
}

__global__ __launch_bounds__(256) void invdeg_kernel(float* __restrict__ deg, int n) {
  int i = blockIdx.x * blockDim.x + threadIdx.x;
  if (i < n) deg[i] = 1.0f / fmaxf(deg[i], 1.0f);
}

// acc[dst] += P[src] * invdeg[dst]; one wave per edge, 4 floats per lane
__global__ __launch_bounds__(256) void edge_agg_kernel(
    const float* __restrict__ P, const int* __restrict__ src, const int* __restrict__ dst,
    int E, float* __restrict__ acc, const float* __restrict__ invdeg) {
  long long t = (long long)blockIdx.x * blockDim.x + threadIdx.x;
  long long total = (long long)E * 32;
  if (t >= total) return;
  int e = (int)(t >> 5), l = (int)(t & 31);
  int s = src[e], d = dst[e];
  float w = invdeg[d];
  const float4 v = *(const float4*)(P + (size_t)s * H + l * 4);
  float* o = acc + (size_t)d * H + l * 4;
  atomicAdd(o + 0, v.x * w);
  atomicAdd(o + 1, v.y * w);
  atomicAdd(o + 2, v.z * w);
  atomicAdd(o + 3, v.w * w);
}

__global__ __launch_bounds__(256) void relu_kernel(float* __restrict__ x, long long n) {
  long long i = (long long)blockIdx.x * blockDim.x + threadIdx.x;
  if (i < n) x[i] = fmaxf(x[i], 0.0f);
}

// ---------------------------------------------------------------------------
// Query injection
// ---------------------------------------------------------------------------
__global__ void qnorm_kernel(const float* __restrict__ q, float* __restrict__ qstat) {
  __shared__ float red[256];
  float s = 0.f;
  for (int k = threadIdx.x; k < 512; k += 256) { float v = q[k]; s += v * v; }
  red[threadIdx.x] = s;
  __syncthreads();
  for (int off = 128; off > 0; off >>= 1) {
    if ((int)threadIdx.x < off) red[threadIdx.x] += red[threadIdx.x + off];
    __syncthreads();
  }
  if (threadIdx.x == 0) qstat[0] = fmaxf(sqrtf(red[0]), 1e-12f);
}

__global__ void qv_kernel(const float* __restrict__ q, const float* __restrict__ Wq,
                          const float* __restrict__ bq, float* __restrict__ qv) {
  int h = threadIdx.x;  // 128 threads
  float s = 0.f;
  for (int k = 0; k < 512; ++k) s += q[k] * Wq[(size_t)k * H + h];
  qv[h] = s + bq[h];
}

// x0 (= tab2, in place) += max(cos_sim(x_i, q),0-clamped) * qv ; wave per row
__global__ __launch_bounds__(256) void inject_kernel(
    const float* __restrict__ tx, const float* __restrict__ q,
    const float* __restrict__ qstat, const float* __restrict__ qv,
    float* __restrict__ x0, int nT) {
  int wv = (int)((blockIdx.x * blockDim.x + threadIdx.x) >> 5);
  int lane = threadIdx.x & 31;
  if (wv >= nT) return;
  const float* xr = tx + (size_t)wv * 512;
  float xq = 0.f, xx = 0.f;
  for (int k = lane; k < 512; k += 32) {
    float v = xr[k];
    xq += v * q[k];
    xx += v * v;
  }
  for (int m = 16; m > 0; m >>= 1) {
    xq += __shfl_xor(xq, m, 32);
    xx += __shfl_xor(xx, m, 32);
  }
  float w = fmaxf(xq, 0.f) / (fmaxf(sqrtf(xx), 1e-12f) * qstat[0]);
  float* o = x0 + (size_t)wv * H;
  for (int c = lane; c < H; c += 32) o[c] += w * qv[c];
}

// ---------------------------------------------------------------------------
// APPNP
// ---------------------------------------------------------------------------
__global__ __launch_bounds__(256) void degw_kernel(const int* __restrict__ dst,
                                                   const float* __restrict__ ew,
                                                   int E, float* __restrict__ deg) {
  int e = blockIdx.x * blockDim.x + threadIdx.x;
  if (e < E) atomicAdd(&deg[dst[e]], ew[e]);
}

__global__ __launch_bounds__(256) void dis_kernel(const float* __restrict__ deg,
                                                  float* __restrict__ dis, int n) {
  int i = blockIdx.x * blockDim.x + threadIdx.x;
  if (i < n) dis[i] = rsqrtf(deg[i] + 1.0f);  // +1 = self-loop weight
}

__global__ __launch_bounds__(256) void enorm_kernel(
    const int* __restrict__ src, const int* __restrict__ dst,
    const float* __restrict__ ew, const float* __restrict__ dis,
    int E, float* __restrict__ en) {
  int e = blockIdx.x * blockDim.x + threadIdx.x;
  if (e < E) en[e] = dis[src[e]] * ew[e] * dis[dst[e]];
}

// zN = ALPHA*x0 + (1-ALPHA)*dis[i]^2*zP   (self-loop term + teleport)
__global__ __launch_bounds__(256) void appnp_self_kernel(
    const float* __restrict__ x0, const float* __restrict__ zP,
    const float* __restrict__ dis, float* __restrict__ zN, int n) {
  long long t = (long long)blockIdx.x * blockDim.x + threadIdx.x;
  if (t >= (long long)n * 32) return;
  int i = (int)(t >> 5), l = (int)(t & 31);
  float dd = dis[i];
  dd = (1.0f - ALPHA) * dd * dd;
  const float4 x = *(const float4*)(x0 + (size_t)i * H + l * 4);
  const float4 z = *(const float4*)(zP + (size_t)i * H + l * 4);
  float4 o;
  o.x = ALPHA * x.x + dd * z.x;
  o.y = ALPHA * x.y + dd * z.y;
  o.z = ALPHA * x.z + dd * z.z;
  o.w = ALPHA * x.w + dd * z.w;
  *(float4*)(zN + (size_t)i * H + l * 4) = o;
}

__global__ __launch_bounds__(256) void appnp_edge_kernel(
    const float* __restrict__ zP, const int* __restrict__ src, const int* __restrict__ dst,
    const float* __restrict__ en, int E, float* __restrict__ zN) {
  long long t = (long long)blockIdx.x * blockDim.x + threadIdx.x;
  if (t >= (long long)E * 32) return;
  int e = (int)(t >> 5), l = (int)(t & 31);
  int s = src[e], d = dst[e];
  float w = (1.0f - ALPHA) * en[e];
  const float4 v = *(const float4*)(zP + (size_t)s * H + l * 4);
  float* o = zN + (size_t)d * H + l * 4;
  atomicAdd(o + 0, v.x * w);
  atomicAdd(o + 1, v.y * w);
  atomicAdd(o + 2, v.z * w);
  atomicAdd(o + 3, v.w * w);
}

// out[i] = z[i,:] . Wlin + blin ; wave per row
__global__ __launch_bounds__(256) void output_kernel(
    const float* __restrict__ z, const float* __restrict__ wlin,
    const float* __restrict__ blin, float* __restrict__ out, int nT) {
  int wv = (int)((blockIdx.x * blockDim.x + threadIdx.x) >> 5);
  int lane = threadIdx.x & 31;
  if (wv >= nT) return;
  float s = 0.f;
  for (int c = lane; c < H; c += 32) s += z[(size_t)wv * H + c] * wlin[c];
  for (int m = 16; m > 0; m >>= 1) s += __shfl_xor(s, m, 32);
  if (lane == 0) out[wv] = s + blin[0];
}

// ---------------------------------------------------------------------------
// Host
// ---------------------------------------------------------------------------
static inline void gemm128(const float* A, const __bf16* Wt, float* C, int M, int K,
                           const float* bias, hipStream_t s) {
  gemm_wmma_kernel<<<M / 32, 256, 0, s>>>(A, Wt, C, M, K, bias);
}

extern "C" void kernel_launch(void* const* d_in, const int* in_sizes, int n_in,
                              void* d_out, int out_size, void* d_ws, size_t ws_size,
                              hipStream_t stream) {
  (void)in_sizes; (void)n_in; (void)out_size; (void)ws_size;
  const float* table_x  = (const float*)d_in[0];
  const float* column_x = (const float*)d_in[1];
  const float* q        = (const float*)d_in[2];
  const float* Wl1_tc  = (const float*)d_in[3];
  const float* bl1_tc  = (const float*)d_in[4];
  const float* Wr1_tc  = (const float*)d_in[5];
  const float* Wl1_rev = (const float*)d_in[6];
  const float* bl1_rev = (const float*)d_in[7];
  const float* Wr1_rev = (const float*)d_in[8];
  const float* Wl1_cs  = (const float*)d_in[9];
  const float* bl1_cs  = (const float*)d_in[10];
  const float* Wr1_cs  = (const float*)d_in[11];
  const float* Wl1_ns  = (const float*)d_in[12];
  const float* bl1_ns  = (const float*)d_in[13];
  const float* Wr1_ns  = (const float*)d_in[14];
  const float* Wl1_ds  = (const float*)d_in[15];
  const float* bl1_ds  = (const float*)d_in[16];
  const float* Wr1_ds  = (const float*)d_in[17];
  const float* Wl1_ts  = (const float*)d_in[18];
  const float* bl1_ts  = (const float*)d_in[19];
  const float* Wr1_ts  = (const float*)d_in[20];
  const float* Wl2_rev = (const float*)d_in[21];
  const float* bl2_rev = (const float*)d_in[22];
  const float* Wr2_rev = (const float*)d_in[23];
  const float* Wl2_ts  = (const float*)d_in[24];
  const float* bl2_ts  = (const float*)d_in[25];
  const float* Wr2_ts  = (const float*)d_in[26];
  const float* Wq      = (const float*)d_in[27];
  const float* bq      = (const float*)d_in[28];
  const float* Wlin    = (const float*)d_in[29];
  const float* blin    = (const float*)d_in[30];
  const float* ts_ew   = (const float*)d_in[31];
  const int* tc_src = (const int*)d_in[32];
  const int* tc_dst = (const int*)d_in[33];
  const int* cs_ei  = (const int*)d_in[34];
  const int* ns_ei  = (const int*)d_in[35];
  const int* ds_ei  = (const int*)d_in[36];
  const int* ts_ei  = (const int*)d_in[37];
  const int* cs_s = cs_ei,            *cs_d = cs_ei + E_SIM;
  const int* ns_s = ns_ei,            *ns_d = ns_ei + E_SIM;
  const int* ds_s = ds_ei,            *ds_d = ds_ei + E_SIM;
  const int* ts_s = ts_ei,            *ts_d = ts_ei + E_TS;

  // ---- workspace carve-up (256B aligned) ----
  size_t off = 0;
  char* base = (char*)d_ws;
  auto alloc = [&](size_t bytes) -> char* {
    char* p = base + off;
    off += (bytes + 255) & ~(size_t)255;
    return p;
  };
  __bf16* wt_l1tc  = (__bf16*)alloc((size_t)512 * H * 2);
  __bf16* wt_l1cs  = (__bf16*)alloc((size_t)256 * H * 2);
  __bf16* wt_l1ns  = (__bf16*)alloc((size_t)256 * H * 2);
  __bf16* wt_l1ds  = (__bf16*)alloc((size_t)256 * H * 2);
  __bf16* wt_l1rev = (__bf16*)alloc((size_t)256 * H * 2);
  __bf16* wt_l1ts  = (__bf16*)alloc((size_t)512 * H * 2);
  __bf16* wt_rcol  = (__bf16*)alloc((size_t)256 * H * 2);
  __bf16* wt_rtab  = (__bf16*)alloc((size_t)512 * H * 2);
  __bf16* wt_l2rev = (__bf16*)alloc((size_t)128 * H * 2);
  __bf16* wt_l2ts  = (__bf16*)alloc((size_t)128 * H * 2);
  __bf16* wt_r2    = (__bf16*)alloc((size_t)128 * H * 2);
  float* b_col  = (float*)alloc(H * 4);
  float* b_tab  = (float*)alloc(H * 4);
  float* b_t2   = (float*)alloc(H * 4);
  float* qv     = (float*)alloc(H * 4);
  float* qstat  = (float*)alloc(4);
  float* bufC   = (float*)alloc((size_t)NC * H * 4);
  float* colAcc = (float*)alloc((size_t)NC * H * 4);
  float* bufT   = (float*)alloc((size_t)NT * H * 4);
  float* tabAcc = (float*)alloc((size_t)NT * H * 4);
  float* tab2   = (float*)alloc((size_t)NT * H * 4);
  float* zA     = (float*)alloc((size_t)NT * H * 4);
  float* zB     = (float*)alloc((size_t)NT * H * 4);
  float* degC   = (float*)alloc((size_t)NC * 4);
  float* degT   = (float*)alloc((size_t)NT * 4);
  float* disT   = (float*)alloc((size_t)NT * 4);
  float* enorm  = (float*)alloc((size_t)E_TS * 4);

  auto blocksFor = [](long long n) -> unsigned { return (unsigned)((n + 255) / 256); };

  // ---- weight / bias prep (bf16 transposed; Wr & bias summed per dst type) ----
  wtprep_kernel<<<blocksFor((long long)512 * H), 256, 0, stream>>>(Wl1_tc, nullptr, nullptr, nullptr, 512, wt_l1tc);
  wtprep_kernel<<<blocksFor((long long)256 * H), 256, 0, stream>>>(Wl1_cs, nullptr, nullptr, nullptr, 256, wt_l1cs);
  wtprep_kernel<<<blocksFor((long long)256 * H), 256, 0, stream>>>(Wl1_ns, nullptr, nullptr, nullptr, 256, wt_l1ns);
  wtprep_kernel<<<blocksFor((long long)256 * H), 256, 0, stream>>>(Wl1_ds, nullptr, nullptr, nullptr, 256, wt_l1ds);
  wtprep_kernel<<<blocksFor((long long)256 * H), 256, 0, stream>>>(Wl1_rev, nullptr, nullptr, nullptr, 256, wt_l1rev);
  wtprep_kernel<<<blocksFor((long long)512 * H), 256, 0, stream>>>(Wl1_ts, nullptr, nullptr, nullptr, 512, wt_l1ts);
  wtprep_kernel<<<blocksFor((long long)256 * H), 256, 0, stream>>>(Wr1_tc, Wr1_cs, Wr1_ns, Wr1_ds, 256, wt_rcol);
  wtprep_kernel<<<blocksFor((long long)512 * H), 256, 0, stream>>>(Wr1_rev, Wr1_ts, nullptr, nullptr, 512, wt_rtab);
  wtprep_kernel<<<blocksFor((long long)128 * H), 256, 0, stream>>>(Wl2_rev, nullptr, nullptr, nullptr, 128, wt_l2rev);
  wtprep_kernel<<<blocksFor((long long)128 * H), 256, 0, stream>>>(Wl2_ts, nullptr, nullptr, nullptr, 128, wt_l2ts);
  wtprep_kernel<<<blocksFor((long long)128 * H), 256, 0, stream>>>(Wr2_rev, Wr2_ts, nullptr, nullptr, 128, wt_r2);
  bias4_kernel<<<1, 128, 0, stream>>>(bl1_tc, bl1_cs, bl1_ns, bl1_ds, b_col);
  bias4_kernel<<<1, 128, 0, stream>>>(bl1_rev, bl1_ts, nullptr, nullptr, b_tab);
  bias4_kernel<<<1, 128, 0, stream>>>(bl2_rev, bl2_ts, nullptr, nullptr, b_t2);

  // ---- layer 1 : column destination ----
  gemm128(column_x, wt_rcol, colAcc, NC, 256, b_col, stream);
  // tc -> col
  hipMemsetAsync(degC, 0, (size_t)NC * 4, stream);
  count_deg_kernel<<<blocksFor(E_TC), 256, 0, stream>>>(tc_dst, E_TC, degC);
  invdeg_kernel<<<blocksFor(NC), 256, 0, stream>>>(degC, NC);
  gemm128(table_x, wt_l1tc, bufT, NT, 512, nullptr, stream);
  edge_agg_kernel<<<blocksFor((long long)E_TC * 32), 256, 0, stream>>>(bufT, tc_src, tc_dst, E_TC, colAcc, degC);
  // cs / ns / ds -> col
  hipMemsetAsync(degC, 0, (size_t)NC * 4, stream);
  count_deg_kernel<<<blocksFor(E_SIM), 256, 0, stream>>>(cs_d, E_SIM, degC);
  invdeg_kernel<<<blocksFor(NC), 256, 0, stream>>>(degC, NC);
  gemm128(column_x, wt_l1cs, bufC, NC, 256, nullptr, stream);
  edge_agg_kernel<<<blocksFor((long long)E_SIM * 32), 256, 0, stream>>>(bufC, cs_s, cs_d, E_SIM, colAcc, degC);

  hipMemsetAsync(degC, 0, (size_t)NC * 4, stream);
  count_deg_kernel<<<blocksFor(E_SIM), 256, 0, stream>>>(ns_d, E_SIM, degC);
  invdeg_kernel<<<blocksFor(NC), 256, 0, stream>>>(degC, NC);
  gemm128(column_x, wt_l1ns, bufC, NC, 256, nullptr, stream);
  edge_agg_kernel<<<blocksFor((long long)E_SIM * 32), 256, 0, stream>>>(bufC, ns_s, ns_d, E_SIM, colAcc, degC);

  hipMemsetAsync(degC, 0, (size_t)NC * 4, stream);
  count_deg_kernel<<<blocksFor(E_SIM), 256, 0, stream>>>(ds_d, E_SIM, degC);
  invdeg_kernel<<<blocksFor(NC), 256, 0, stream>>>(degC, NC);
  gemm128(column_x, wt_l1ds, bufC, NC, 256, nullptr, stream);
  edge_agg_kernel<<<blocksFor((long long)E_SIM * 32), 256, 0, stream>>>(bufC, ds_s, ds_d, E_SIM, colAcc, degC);

  relu_kernel<<<blocksFor((long long)NC * H), 256, 0, stream>>>(colAcc, (long long)NC * H);

  // ---- layer 1 : table destination ----
  gemm128(table_x, wt_rtab, tabAcc, NT, 512, b_tab, stream);
  // rev (col -> tab over reversed tc edges)
  hipMemsetAsync(degT, 0, (size_t)NT * 4, stream);
  count_deg_kernel<<<blocksFor(E_TC), 256, 0, stream>>>(tc_src, E_TC, degT);
  invdeg_kernel<<<blocksFor(NT), 256, 0, stream>>>(degT, NT);
  gemm128(column_x, wt_l1rev, bufC, NC, 256, nullptr, stream);
  edge_agg_kernel<<<blocksFor((long long)E_TC * 32), 256, 0, stream>>>(bufC, tc_dst, tc_src, E_TC, tabAcc, degT);
  // ts (tab -> tab)
  hipMemsetAsync(degT, 0, (size_t)NT * 4, stream);
  count_deg_kernel<<<blocksFor(E_TS), 256, 0, stream>>>(ts_d, E_TS, degT);
  invdeg_kernel<<<blocksFor(NT), 256, 0, stream>>>(degT, NT);
  gemm128(table_x, wt_l1ts, bufT, NT, 512, nullptr, stream);
  edge_agg_kernel<<<blocksFor((long long)E_TS * 32), 256, 0, stream>>>(bufT, ts_s, ts_d, E_TS, tabAcc, degT);

  relu_kernel<<<blocksFor((long long)NT * H), 256, 0, stream>>>(tabAcc, (long long)NT * H);

  // ---- layer 2 (table output only) ----
  gemm128(tabAcc, wt_r2, tab2, NT, 128, b_t2, stream);
  hipMemsetAsync(degT, 0, (size_t)NT * 4, stream);
  count_deg_kernel<<<blocksFor(E_TC), 256, 0, stream>>>(tc_src, E_TC, degT);
  invdeg_kernel<<<blocksFor(NT), 256, 0, stream>>>(degT, NT);
  gemm128(colAcc, wt_l2rev, bufC, NC, 128, nullptr, stream);
  edge_agg_kernel<<<blocksFor((long long)E_TC * 32), 256, 0, stream>>>(bufC, tc_dst, tc_src, E_TC, tab2, degT);

  hipMemsetAsync(degT, 0, (size_t)NT * 4, stream);
  count_deg_kernel<<<blocksFor(E_TS), 256, 0, stream>>>(ts_d, E_TS, degT);
  invdeg_kernel<<<blocksFor(NT), 256, 0, stream>>>(degT, NT);
  gemm128(tabAcc, wt_l2ts, bufT, NT, 128, nullptr, stream);
  edge_agg_kernel<<<blocksFor((long long)E_TS * 32), 256, 0, stream>>>(bufT, ts_s, ts_d, E_TS, tab2, degT);

  // ---- retrieval-weighted query injection: x0 = tab2 + wgt*qv (in place) ----
  qnorm_kernel<<<1, 256, 0, stream>>>(q, qstat);
  qv_kernel<<<1, 128, 0, stream>>>(q, Wq, bq, qv);
  inject_kernel<<<blocksFor((long long)NT * 32), 256, 0, stream>>>(table_x, q, qstat, qv, tab2, NT);

  // ---- APPNP ----
  hipMemsetAsync(degT, 0, (size_t)NT * 4, stream);
  degw_kernel<<<blocksFor(E_TS), 256, 0, stream>>>(ts_d, ts_ew, E_TS, degT);
  dis_kernel<<<blocksFor(NT), 256, 0, stream>>>(degT, disT, NT);
  enorm_kernel<<<blocksFor(E_TS), 256, 0, stream>>>(ts_s, ts_d, ts_ew, disT, E_TS, enorm);

  const float* zP = tab2;  // z_0 = x0
  float* zN = zA;
  for (int it = 0; it < K_APPNP; ++it) {
    zN = (it & 1) ? zB : zA;
    appnp_self_kernel<<<blocksFor((long long)NT * 32), 256, 0, stream>>>(tab2, zP, disT, zN, NT);
    appnp_edge_kernel<<<blocksFor((long long)E_TS * 32), 256, 0, stream>>>(zP, ts_s, ts_d, enorm, E_TS, zN);
    zP = zN;
  }

  output_kernel<<<blocksFor((long long)NT * 32), 256, 0, stream>>>(zP, Wlin, blin, (float*)d_out, NT);
}
